// TGAT_40535901340473
// MI455X (gfx1250) — compile-verified, matrix operand order checked by reading, not run on hardware
//
#include <hip/hip_runtime.h>
#include <cstdint>

typedef __bf16 bf16t;
typedef __attribute__((ext_vector_type(16))) __bf16 v16bf;
typedef __attribute__((ext_vector_type(8)))  float  v8f;

// ---------------------------------------------------------------------------
// CDNA5 WMMA helpers (wave32, 16x16x32 bf16 -> f32)
// ---------------------------------------------------------------------------
__device__ __forceinline__ v8f wmma_bf16(v16bf a, v16bf b, v8f c) {
  return __builtin_amdgcn_wmma_f32_16x16x32_bf16(false, a, false, b, (short)0, c,
                                                 false, false);
}

// Build one lane's 16x32 bf16 A/B fragment slice from a row-major f32 row.
// `rowp` points at the first element of this row's 32-wide K-slice.
// Per ISA layout: element e holds K = (e<8 ? 8*sub+e : 16+8*sub+(e-8)),
// where sub = lane>=16. Two contiguous 8-float chunks per lane.
__device__ __forceinline__ v16bf frag_f32(const float* __restrict__ rowp, int sub) {
  v16bf f;
  const float* p0 = rowp + 8 * sub;
  const float* p1 = rowp + 16 + 8 * sub;
#pragma unroll
  for (int e = 0; e < 8; ++e) {
    f[e]     = (bf16t)p0[e];
    f[e + 8] = (bf16t)p1[e];
  }
  return f;
}

// float <-> monotone int key (for float atomicMax via integer atomics)
__device__ __forceinline__ int fkey(float f) {
  int i = __float_as_int(f);
  return i >= 0 ? i : (i ^ 0x7fffffff);
}
__device__ __forceinline__ float fdekey(int k) {
  return __int_as_float(k >= 0 ? k : (k ^ 0x7fffffff));
}

__device__ __forceinline__ void atomAddF(float* p, float v) {
  __hip_atomic_fetch_add(p, v, __ATOMIC_RELAXED, __HIP_MEMORY_SCOPE_AGENT);
}

// ---------------------------------------------------------------------------
// K0: init segment-max keys (-inf), denom, agg
// ---------------------------------------------------------------------------
__global__ __launch_bounds__(256) void init_kernel(int* __restrict__ mkey,
                                                   float* __restrict__ denom,
                                                   float* __restrict__ agg, int N) {
  int i = blockIdx.x * blockDim.x + threadIdx.x;
  if (i < N * 2) {
    mkey[i]  = fkey(-__builtin_inff());
    denom[i] = 0.0f;
  }
  if (i < N * 32) agg[i] = 0.0f;
}

// ---------------------------------------------------------------------------
// K1: h = relu(x @ W_lin^T + b_lin); q/k/v/s = h @ W*^T + b*   (all WMMA)
// One wave = one 16-row tile. Block = 8 waves = 128 rows.
// ---------------------------------------------------------------------------
__global__ __launch_bounds__(256) void node_mlp_kernel(
    const float* __restrict__ x,
    const float* __restrict__ W_lin, const float* __restrict__ b_lin,
    const float* __restrict__ Wq, const float* __restrict__ bq,
    const float* __restrict__ Wk, const float* __restrict__ bk,
    const float* __restrict__ Wv, const float* __restrict__ bv,
    const float* __restrict__ Ws, const float* __restrict__ bs,
    float* __restrict__ q, float* __restrict__ kn,
    float* __restrict__ vn, float* __restrict__ sb, int N) {
  __shared__ float hstage[8][16][32];
  const int wave = threadIdx.x >> 5;
  const int lane = threadIdx.x & 31;
  const int sub  = lane >> 4;
  const int r    = lane & 15;
  const int row0 = (blockIdx.x * 8 + wave) * 16;
  if (row0 >= N) return;  // wave-uniform; EXEC stays all-ones for WMMA

  const int rowA = min(row0 + r, N - 1);

  // Stage 1: h tile (16 x 32) = relu(x[16x128] @ W_lin^T + b)
  v8f c0 = {}, c1 = {};
#pragma unroll
  for (int k0 = 0; k0 < 128; k0 += 32) {
    v16bf a  = frag_f32(x + (size_t)rowA * 128 + k0, sub);
    v16bf b0 = frag_f32(W_lin + (size_t)r * 128 + k0, sub);        // cols 0..15
    v16bf b1 = frag_f32(W_lin + (size_t)(16 + r) * 128 + k0, sub); // cols 16..31
    c0 = wmma_bf16(a, b0, c0);
    c1 = wmma_bf16(a, b1, c1);
  }
  const float bl0 = b_lin[r], bl1 = b_lin[16 + r];
  float* hs = &hstage[wave][0][0];
#pragma unroll
  for (int i = 0; i < 8; ++i) {
    const int m = i + 8 * sub;  // C layout: VGPR i -> row m, lane -> col
    hs[m * 32 + r]      = fmaxf(c0[i] + bl0, 0.0f);
    hs[m * 32 + 16 + r] = fmaxf(c1[i] + bl1, 0.0f);
  }
  // wave-local LDS staging; DS pipe is in-order per wave
  asm volatile("s_wait_dscnt 0" ::: "memory");
  const v16bf a2 = frag_f32(hs + r * 32, sub);  // h tile in A layout

  // Stage 2: four 32x32 projections from h
  const float* Wt[4] = {Wq, Wk, Wv, Ws};
  const float* Bt[4] = {bq, bk, bv, bs};
  float*       Ot[4] = {q, kn, vn, sb};
#pragma unroll
  for (int t = 0; t < 4; ++t) {
    v16bf b0 = frag_f32(Wt[t] + r * 32, sub);
    v16bf b1 = frag_f32(Wt[t] + (16 + r) * 32, sub);
    v8f d0 = {}, d1 = {};
    d0 = wmma_bf16(a2, b0, d0);
    d1 = wmma_bf16(a2, b1, d1);
    const float bb0 = Bt[t][r], bb1 = Bt[t][16 + r];
#pragma unroll
    for (int i = 0; i < 8; ++i) {
      const int m = i + 8 * sub;
      if (row0 + m < N) {
        Ot[t][(size_t)(row0 + m) * 32 + r]      = d0[i] + bb0;
        Ot[t][(size_t)(row0 + m) * 32 + 16 + r] = d1[i] + bb1;
      }
    }
  }
}

// ---------------------------------------------------------------------------
// Shared per-edge-tile encoder: time/degree cos-encoding -> e_attr -> e_proj
// Returns e_proj C-fragments (cols r / 16+r), bias `be` NOT yet added.
// ---------------------------------------------------------------------------
struct EdgeProj { v8f ep0, ep1; };

__device__ __forceinline__ EdgeProj edge_proj(
    int e0, int E, int sub, int r, float* stage,
    const int* __restrict__ src, const int* __restrict__ dst,
    const float* __restrict__ node_time, const float* __restrict__ edge_time,
    const float* __restrict__ deg,
    const float* __restrict__ W_time, const float* __restrict__ b_time,
    const float* __restrict__ W_deg, const float* __restrict__ b_deg,
    const float* __restrict__ W_comb, const float* __restrict__ b_comb,
    const float* __restrict__ We) {
  const int er  = min(e0 + r, E - 1);
  const int s_r = src[er], d_r = dst[er];
  const float rt = node_time[s_r] - edge_time[er];
  const float rd = deg[s_r] - deg[d_r];

  // enc[E x 64] built directly in A-fragment layout; cols 0..31 = t_enc,
  // 32..63 = d_enc -> two K-steps of 32.
  v16bf aT, aD;
  const int kb0 = 8 * sub, kb1 = 16 + 8 * sub;
#pragma unroll
  for (int e = 0; e < 8; ++e) {
    const int j0 = kb0 + e, j1 = kb1 + e;
    aT[e]     = (bf16t)__cosf(rt * W_time[j0] + b_time[j0]);
    aT[e + 8] = (bf16t)__cosf(rt * W_time[j1] + b_time[j1]);
    aD[e]     = (bf16t)__cosf(rd * W_deg[j0] + b_deg[j0]);
    aD[e + 8] = (bf16t)__cosf(rd * W_deg[j1] + b_deg[j1]);
  }

  // e_attr = enc @ W_comb^T + b_comb   (K=64 -> 2 steps, N=32 -> 2 tiles)
  v8f c0 = {}, c1 = {};
  {
    v16bf b00 = frag_f32(W_comb + r * 64, sub);
    v16bf b01 = frag_f32(W_comb + (16 + r) * 64, sub);
    c0 = wmma_bf16(aT, b00, c0);
    c1 = wmma_bf16(aT, b01, c1);
    v16bf b10 = frag_f32(W_comb + r * 64 + 32, sub);
    v16bf b11 = frag_f32(W_comb + (16 + r) * 64 + 32, sub);
    c0 = wmma_bf16(aD, b10, c0);
    c1 = wmma_bf16(aD, b11, c1);
  }
  const float bc0 = b_comb[r], bc1 = b_comb[16 + r];
#pragma unroll
  for (int i = 0; i < 8; ++i) {
    const int m = i + 8 * sub;
    stage[m * 32 + r]      = c0[i] + bc0;
    stage[m * 32 + 16 + r] = c1[i] + bc1;
  }
  asm volatile("s_wait_dscnt 0" ::: "memory");
  const v16bf a2 = frag_f32(stage + r * 32, sub);

  // e_proj = e_attr @ We^T
  EdgeProj out;
  v16bf b0 = frag_f32(We + r * 32, sub);
  v16bf b1 = frag_f32(We + (16 + r) * 32, sub);
  v8f d0 = {}, d1 = {};
  out.ep0 = wmma_bf16(a2, b0, d0);
  out.ep1 = wmma_bf16(a2, b1, d1);
  return out;
}

// ---------------------------------------------------------------------------
// K2: per-edge attention logits + segment-max (float atomicMax via int keys)
// One wave = 16 edges.
// ---------------------------------------------------------------------------
__global__ __launch_bounds__(256) void edge_logits_kernel(
    const int* __restrict__ ei, int E,
    const float* __restrict__ node_time, const float* __restrict__ edge_time,
    const float* __restrict__ deg,
    const float* __restrict__ W_time, const float* __restrict__ b_time,
    const float* __restrict__ W_deg, const float* __restrict__ b_deg,
    const float* __restrict__ W_comb, const float* __restrict__ b_comb,
    const float* __restrict__ We, const float* __restrict__ be,
    const float* __restrict__ qn, const float* __restrict__ kn,
    float* __restrict__ logits, int* __restrict__ mkey) {
  __shared__ float estage[8][16][32];
  const int wave = threadIdx.x >> 5;
  const int lane = threadIdx.x & 31;
  const int sub  = lane >> 4;
  const int r    = lane & 15;
  const int e0   = (blockIdx.x * 8 + wave) * 16;
  if (e0 >= E) return;
  const int* src = ei;
  const int* dst = ei + E;

  EdgeProj P = edge_proj(e0, E, sub, r, &estage[wave][0][0], src, dst,
                         node_time, edge_time, deg, W_time, b_time, W_deg,
                         b_deg, W_comb, b_comb, We);

  int se[8], de[8];
#pragma unroll
  for (int i = 0; i < 8; ++i) {
    const int e = min(e0 + i + 8 * sub, E - 1);
    se[i] = src[e];
    de[i] = dst[e];
  }

  const float be0 = be[r], be1 = be[16 + r];
  float p0[8], p1[8];
#pragma unroll
  for (int i = 0; i < 8; ++i) {
    const float k0 = kn[(size_t)se[i] * 32 + r]      + P.ep0[i] + be0;
    const float k1 = kn[(size_t)se[i] * 32 + 16 + r] + P.ep1[i] + be1;
    p0[i] = qn[(size_t)de[i] * 32 + r]      * k0;   // head 0, col r
    p1[i] = qn[(size_t)de[i] * 32 + 16 + r] * k1;   // head 1, col 16+r
  }
  // reduce the 16 columns of each head across the 16-lane half
#pragma unroll
  for (int mask = 8; mask >= 1; mask >>= 1) {
#pragma unroll
    for (int i = 0; i < 8; ++i) {
      p0[i] += __shfl_xor(p0[i], mask, 32);
      p1[i] += __shfl_xor(p1[i], mask, 32);
    }
  }
  if (r == 0) {  // lanes 0 (rows 0..7) and 16 (rows 8..15)
#pragma unroll
    for (int i = 0; i < 8; ++i) {
      const int e = e0 + i + 8 * sub;
      if (e < E) {
        const float l0 = p0[i] * 0.25f;  // 1/sqrt(16)
        const float l1 = p1[i] * 0.25f;
        logits[(size_t)e * 2 + 0] = l0;
        logits[(size_t)e * 2 + 1] = l1;
        atomicMax(&mkey[de[i] * 2 + 0], fkey(l0));
        atomicMax(&mkey[de[i] * 2 + 1], fkey(l1));
      }
    }
  }
}

// ---------------------------------------------------------------------------
// K3: alpha = exp(logit - max), scatter alpha and alpha*v with f32 atomics
// ---------------------------------------------------------------------------
__global__ __launch_bounds__(256) void edge_accum_kernel(
    const int* __restrict__ ei, int E,
    const float* __restrict__ node_time, const float* __restrict__ edge_time,
    const float* __restrict__ deg,
    const float* __restrict__ W_time, const float* __restrict__ b_time,
    const float* __restrict__ W_deg, const float* __restrict__ b_deg,
    const float* __restrict__ W_comb, const float* __restrict__ b_comb,
    const float* __restrict__ We, const float* __restrict__ be,
    const float* __restrict__ vn, const float* __restrict__ logits,
    const int* __restrict__ mkey,
    float* __restrict__ agg, float* __restrict__ denom) {
  __shared__ float estage[8][16][32];
  const int wave = threadIdx.x >> 5;
  const int lane = threadIdx.x & 31;
  const int sub  = lane >> 4;
  const int r    = lane & 15;
  const int e0   = (blockIdx.x * 8 + wave) * 16;
  if (e0 >= E) return;
  const int* src = ei;
  const int* dst = ei + E;

  EdgeProj P = edge_proj(e0, E, sub, r, &estage[wave][0][0], src, dst,
                         node_time, edge_time, deg, W_time, b_time, W_deg,
                         b_deg, W_comb, b_comb, We);

  const float be0 = be[r], be1 = be[16 + r];
#pragma unroll
  for (int i = 0; i < 8; ++i) {
    const int e = e0 + i + 8 * sub;
    if (e >= E) continue;  // divergence only after all WMMAs
    const int s = src[e], d = dst[e];
    const float a0 = __expf(logits[(size_t)e * 2 + 0] - fdekey(mkey[d * 2 + 0]));
    const float a1 = __expf(logits[(size_t)e * 2 + 1] - fdekey(mkey[d * 2 + 1]));
    const float v0 = vn[(size_t)s * 32 + r]      + P.ep0[i] + be0;
    const float v1 = vn[(size_t)s * 32 + 16 + r] + P.ep1[i] + be1;
    atomAddF(&agg[(size_t)d * 32 + r],      a0 * v0);
    atomAddF(&agg[(size_t)d * 32 + 16 + r], a1 * v1);
    if (r == 0) {
      atomAddF(&denom[d * 2 + 0], a0);
      atomAddF(&denom[d * 2 + 1], a1);
    }
  }
}

// ---------------------------------------------------------------------------
// K4: h1 = agg/denom + skip; out = log_softmax(h1 @ W_out^T + b_out)
// ---------------------------------------------------------------------------
__global__ __launch_bounds__(256) void finalize_kernel(
    const float* __restrict__ agg, const float* __restrict__ denom,
    const float* __restrict__ sb, const float* __restrict__ W_out,
    const float* __restrict__ b_out, float* __restrict__ out, int N) {
  const int n = blockIdx.x * blockDim.x + threadIdx.x;
  if (n >= N) return;
  float o0 = b_out[0], o1 = b_out[1];
  const float d0 = denom[n * 2 + 0] + 1e-16f;
  const float d1 = denom[n * 2 + 1] + 1e-16f;
#pragma unroll
  for (int j = 0; j < 32; ++j) {
    const float h1 = agg[(size_t)n * 32 + j] / (j < 16 ? d0 : d1) +
                     sb[(size_t)n * 32 + j];
    o0 += h1 * W_out[j];
    o1 += h1 * W_out[32 + j];
  }
  const float mx  = fmaxf(o0, o1);
  const float lse = mx + __logf(__expf(o0 - mx) + __expf(o1 - mx));
  out[(size_t)n * 2 + 0] = o0 - lse;
  out[(size_t)n * 2 + 1] = o1 - lse;
}

// ---------------------------------------------------------------------------
extern "C" void kernel_launch(void* const* d_in, const int* in_sizes, int n_in,
                              void* d_out, int out_size, void* d_ws,
                              size_t ws_size, hipStream_t stream) {
  const float* x         = (const float*)d_in[0];
  const int*   ei        = (const int*)d_in[1];
  const float* node_time = (const float*)d_in[2];
  const float* edge_time = (const float*)d_in[3];
  const float* deg       = (const float*)d_in[4];
  const float* W_time = (const float*)d_in[5];  const float* b_time = (const float*)d_in[6];
  const float* W_deg  = (const float*)d_in[7];  const float* b_deg  = (const float*)d_in[8];
  const float* W_lin  = (const float*)d_in[9];  const float* b_lin  = (const float*)d_in[10];
  const float* W_comb = (const float*)d_in[11]; const float* b_comb = (const float*)d_in[12];
  const float* Wq = (const float*)d_in[13]; const float* bq = (const float*)d_in[14];
  const float* Wk = (const float*)d_in[15]; const float* bk = (const float*)d_in[16];
  const float* Wv = (const float*)d_in[17]; const float* bv = (const float*)d_in[18];
  const float* We = (const float*)d_in[19]; const float* be = (const float*)d_in[20];
  const float* Ws = (const float*)d_in[21]; const float* bs = (const float*)d_in[22];
  const float* W_out = (const float*)d_in[23]; const float* b_out = (const float*)d_in[24];

  const int N = in_sizes[2];  // node_time is [N]
  const int E = in_sizes[3];  // edge_time is [E,1]

  float* ws = (float*)d_ws;
  size_t o = 0;
  float* q      = ws + o; o += (size_t)N * 32;
  float* kn     = ws + o; o += (size_t)N * 32;
  float* vn     = ws + o; o += (size_t)N * 32;
  float* sb     = ws + o; o += (size_t)N * 32;
  float* logits = ws + o; o += (size_t)E * 2;
  int*   mkey   = (int*)(ws + o); o += (size_t)N * 2;
  float* denom  = ws + o; o += (size_t)N * 2;
  float* agg    = ws + o; o += (size_t)N * 32;

  init_kernel<<<(N * 32 + 255) / 256, 256, 0, stream>>>(mkey, denom, agg, N);

  node_mlp_kernel<<<(N + 127) / 128, 256, 0, stream>>>(
      x, W_lin, b_lin, Wq, bq, Wk, bk, Wv, bv, Ws, bs, q, kn, vn, sb, N);

  const int etiles  = (E + 15) / 16;
  const int eblocks = (etiles + 7) / 8;

  edge_logits_kernel<<<eblocks, 256, 0, stream>>>(
      ei, E, node_time, edge_time, deg, W_time, b_time, W_deg, b_deg, W_comb,
      b_comb, We, be, q, kn, logits, mkey);

  edge_accum_kernel<<<eblocks, 256, 0, stream>>>(
      ei, E, node_time, edge_time, deg, W_time, b_time, W_deg, b_deg, W_comb,
      b_comb, We, be, vn, logits, mkey, agg, denom);

  finalize_kernel<<<(N + 255) / 256, 256, 0, stream>>>(
      agg, denom, sb, W_out, b_out, (float*)d_out, N);
}
